// ConvSwiGLU_27238682591382
// MI455X (gfx1250) — matrix operand order repaired and checked by reading, not compiled, not run on hardware
//
#include <hip/hip_runtime.h>
#include <hip/hip_bf16.h>
#include <cstdint>

#define D_MODEL 1024
#define D_FF    4096
#define BATCH   4
#define SEQ     2048
#define M_TOK   (BATCH * SEQ)   // 8192 tokens

// ---- vector types (ext_vector: trivial, union-safe) ----
typedef __bf16        v16bf __attribute__((ext_vector_type(16)));
typedef __bf16        v8bf  __attribute__((ext_vector_type(8)));
typedef __bf16        v4bf  __attribute__((ext_vector_type(4)));
typedef float         v8f   __attribute__((ext_vector_type(8)));
typedef float         v4f   __attribute__((ext_vector_type(4)));
typedef unsigned int  v4u   __attribute__((ext_vector_type(4)));

union FragAB { v4u u[2]; v16bf v; };   // 32 bytes = 16 bf16

// ---------------------------------------------------------------------------
// 1) elementwise cast f32 -> bf16 (x), 4 elements/thread
// ---------------------------------------------------------------------------
__global__ void cast_f32_bf16_kernel(const float* __restrict__ src,
                                     __bf16* __restrict__ dst, int n4) {
    int i = blockIdx.x * blockDim.x + threadIdx.x;
    if (i >= n4) return;
    v4f s = *(const v4f*)(src + (size_t)i * 4);
    v4bf d;
    d[0] = (__bf16)s[0]; d[1] = (__bf16)s[1];
    d[2] = (__bf16)s[2]; d[3] = (__bf16)s[3];
    *(v4bf*)(dst + (size_t)i * 4) = d;
}

// ---------------------------------------------------------------------------
// 2) transpose + cast: src[R][C] f32 -> dst[C][R] bf16  (LDS tiled 32x32)
// ---------------------------------------------------------------------------
__global__ __launch_bounds__(256)
void transpose_cast_kernel(const float* __restrict__ src,
                           __bf16* __restrict__ dst, int R, int C) {
    __shared__ float tile[32][33];
    int c  = blockIdx.x * 32 + threadIdx.x;      // source col
    int r0 = blockIdx.y * 32;                    // source row base
#pragma unroll
    for (int j = 0; j < 4; ++j) {
        int r = r0 + threadIdx.y + j * 8;
        tile[threadIdx.y + j * 8][threadIdx.x] = src[(size_t)r * C + c];
    }
    __syncthreads();
    int oc  = blockIdx.y * 32 + threadIdx.x;     // dest col  (= source row)
    int or0 = blockIdx.x * 32;                   // dest row base (= source col)
#pragma unroll
    for (int j = 0; j < 4; ++j) {
        int orow = or0 + threadIdx.y + j * 8;
        dst[(size_t)orow * R + oc] = (__bf16)tile[threadIdx.x][threadIdx.y + j * 8];
    }
}

// ---------------------------------------------------------------------------
// 3) WMMA GEMM:  C[M][N] = A[M][K] * BT[N][K]^T + bias[N]
//    A, BT bf16 row-major. Block tile 128x256, 8 waves (2 M x 4 N),
//    wave tile 64x64 => 4x4 accumulators, K-step 32.
//    Inner loop: 16 b128 loads per 16 WMMAs.
// ---------------------------------------------------------------------------
template <bool OUT_BF16>
__global__ __launch_bounds__(256)
void gemm_wmma_bf16_kernel(const __bf16* __restrict__ A,
                           const __bf16* __restrict__ BT,
                           const float*  __restrict__ bias,
                           void* __restrict__ Cout,
                           int M, int N, int K) {
    const int lane  = threadIdx.x & 31;
    const int wid   = threadIdx.x >> 5;
    const int waveN = wid & 3;     // 4 col-blocks of 64
    const int waveM = wid >> 2;    // 2 row-blocks of 64
    const int rowBase = blockIdx.y * 128 + waveM * 64;
    const int colBase = blockIdx.x * 256 + waveN * 64;
    const int half = lane >> 4;    // lane group
    const int l16  = lane & 15;    // m-row for A frag, n-col for B frag / C

    v8f acc[4][4] = {};

    for (int k0 = 0; k0 < K; k0 += 32) {
        FragAB a[4];
#pragma unroll
        for (int mi = 0; mi < 4; ++mi) {
            // A 16x32 bf16 layout: VGPR0-3 <- K = 8*half..8*half+7,
            //                      VGPR4-7 <- K = 16+8*half..16+8*half+7
            const __bf16* ap = A + (size_t)(rowBase + mi * 16 + l16) * K
                                 + k0 + half * 8;
            a[mi].u[0] = *(const v4u*)(ap);
            a[mi].u[1] = *(const v4u*)(ap + 16);
        }
        FragAB b[4];
#pragma unroll
        for (int ni = 0; ni < 4; ++ni) {
            // B 32x16 bf16 layout: VGPR0-3 <- K = 8*half..8*half+7, VGPR4-7 <- K = 16+8*half..16+8*half+7
            const __bf16* bp = BT + (size_t)(colBase + ni * 16 + l16) * K
                                  + k0 + half * 8;
            b[ni].u[0] = *(const v4u*)(bp);
            b[ni].u[1] = *(const v4u*)(bp + 16);
        }
#pragma unroll
        for (int mi = 0; mi < 4; ++mi)
#pragma unroll
            for (int ni = 0; ni < 4; ++ni)
                acc[mi][ni] = __builtin_amdgcn_wmma_f32_16x16x32_bf16(
                    false, a[mi].v, false, b[ni].v,
                    (short)0, acc[mi][ni], false, false);
    }

    // epilogue: bias add + store. D layout: lane l16 = N, VGPR v = M%8, half = M/8
#pragma unroll
    for (int ni = 0; ni < 4; ++ni) {
        const int col = colBase + ni * 16 + l16;
        const float bv = bias[col];
#pragma unroll
        for (int mi = 0; mi < 4; ++mi) {
#pragma unroll
            for (int v = 0; v < 8; ++v) {
                const int row = rowBase + mi * 16 + half * 8 + v;
                const float val = acc[mi][ni][v] + bv;
                if (OUT_BF16)
                    ((__bf16*)Cout)[(size_t)row * N + col] = (__bf16)val;
                else
                    ((float*)Cout)[(size_t)row * N + col] = val;
            }
        }
    }
}

// ---------------------------------------------------------------------------
// 4) depthwise conv (K=5, same padding per batch) + SiLU(gate)*up
//    gate/up: [M_TOK][D_FF] bf16.  8 channels per thread.
// ---------------------------------------------------------------------------
__global__ __launch_bounds__(256)
void conv_silu_mul_kernel(const __bf16* __restrict__ gate,
                          const __bf16* __restrict__ up,
                          const float* __restrict__ cgw,  // [D_FF][1][5]
                          const float* __restrict__ cgb,
                          const float* __restrict__ cuw,
                          const float* __restrict__ cub,
                          __bf16* __restrict__ h) {
    const int F8 = D_FF / 8;  // 512
    int tid = blockIdx.x * blockDim.x + threadIdx.x;
    int f8 = tid % F8;
    int m  = tid / F8;
    if (m >= M_TOK) return;
    const int l  = m & (SEQ - 1);
    const int f0 = f8 * 8;

    float accg[8] = {0.f, 0.f, 0.f, 0.f, 0.f, 0.f, 0.f, 0.f};
    float accu[8] = {0.f, 0.f, 0.f, 0.f, 0.f, 0.f, 0.f, 0.f};
#pragma unroll
    for (int t = 0; t < 5; ++t) {
        int ll = l + t - 2;
        if (ll < 0 || ll >= SEQ) continue;   // zero padding, per-batch
        size_t off = (size_t)(m + t - 2) * D_FF + f0;
        v8bf g = *(const v8bf*)(gate + off);
        v8bf u = *(const v8bf*)(up + off);
#pragma unroll
        for (int j = 0; j < 8; ++j) {
            accg[j] += (float)g[j] * cgw[(size_t)(f0 + j) * 5 + t];
            accu[j] += (float)u[j] * cuw[(size_t)(f0 + j) * 5 + t];
        }
    }
    v8bf hv;
#pragma unroll
    for (int j = 0; j < 8; ++j) {
        float gg = accg[j] + cgb[f0 + j];
        float uu = accu[j] + cub[f0 + j];
        float si = gg / (1.0f + __expf(-gg));   // SiLU
        hv[j] = (__bf16)(si * uu);
    }
    *(v8bf*)(h + (size_t)m * D_FF + f0) = hv;
}

// ---------------------------------------------------------------------------
extern "C" void kernel_launch(void* const* d_in, const int* in_sizes, int n_in,
                              void* d_out, int out_size, void* d_ws, size_t ws_size,
                              hipStream_t stream) {
    const float* x   = (const float*)d_in[0];
    const float* Wg  = (const float*)d_in[1];
    const float* bg  = (const float*)d_in[2];
    const float* Wu  = (const float*)d_in[3];
    const float* bu  = (const float*)d_in[4];
    const float* cgw = (const float*)d_in[5];
    const float* cgb = (const float*)d_in[6];
    const float* cuw = (const float*)d_in[7];
    const float* cub = (const float*)d_in[8];
    const float* Wd  = (const float*)d_in[9];
    const float* bd  = (const float*)d_in[10];
    float* out = (float*)d_out;

    // workspace carve-out (256B aligned)
    char* ws = (char*)d_ws;
    size_t off = 0;
    auto carve = [&](size_t bytes) -> void* {
        void* p = ws + off;
        off = (off + bytes + 255) & ~(size_t)255;
        return p;
    };
    __bf16* xb   = (__bf16*)carve((size_t)M_TOK * D_MODEL * 2);  // 16 MB
    __bf16* WgT  = (__bf16*)carve((size_t)D_FF * D_MODEL * 2);   //  8 MB
    __bf16* WuT  = (__bf16*)carve((size_t)D_FF * D_MODEL * 2);   //  8 MB
    __bf16* WdT  = (__bf16*)carve((size_t)D_MODEL * D_FF * 2);   //  8 MB
    __bf16* gate = (__bf16*)carve((size_t)M_TOK * D_FF * 2);     // 64 MB
    __bf16* up   = (__bf16*)carve((size_t)M_TOK * D_FF * 2);     // 64 MB
    __bf16* hbuf = (__bf16*)carve((size_t)M_TOK * D_FF * 2);     // 64 MB

    // 1) cast x -> bf16
    {
        int n4 = (M_TOK * D_MODEL) / 4;
        cast_f32_bf16_kernel<<<(n4 + 255) / 256, 256, 0, stream>>>(x, xb, n4);
    }
    // 2) transpose + cast weights
    {
        dim3 blk(32, 8);
        dim3 g1(D_FF / 32, D_MODEL / 32);   // Wg [1024][4096] -> WgT [4096][1024]
        transpose_cast_kernel<<<g1, blk, 0, stream>>>(Wg, WgT, D_MODEL, D_FF);
        transpose_cast_kernel<<<g1, blk, 0, stream>>>(Wu, WuT, D_MODEL, D_FF);
        dim3 g2(D_MODEL / 32, D_FF / 32);   // Wd [4096][1024] -> WdT [1024][4096]
        transpose_cast_kernel<<<g2, blk, 0, stream>>>(Wd, WdT, D_FF, D_MODEL);
    }
    // 3) gate = x@Wg + bg ; up = x@Wu + bu   (bf16 out)
    {
        dim3 grid(D_FF / 256, M_TOK / 128);  // (16, 64)
        gemm_wmma_bf16_kernel<true><<<grid, 256, 0, stream>>>(
            xb, WgT, bg, (void*)gate, M_TOK, D_FF, D_MODEL);
        gemm_wmma_bf16_kernel<true><<<grid, 256, 0, stream>>>(
            xb, WuT, bu, (void*)up, M_TOK, D_FF, D_MODEL);
    }
    // 4) depthwise conv + SiLU * up -> h (bf16)
    {
        int total = M_TOK * (D_FF / 8);
        conv_silu_mul_kernel<<<(total + 255) / 256, 256, 0, stream>>>(
            gate, up, cgw, cgb, cuw, cub, hbuf);
    }
    // 5) out = h @ Wd + bd  (f32 out)
    {
        dim3 grid(D_MODEL / 256, M_TOK / 128);  // (4, 64)
        gemm_wmma_bf16_kernel<false><<<grid, 256, 0, stream>>>(
            hbuf, WdT, bd, (void*)out, M_TOK, D_MODEL, D_FF);
    }
    (void)in_sizes; (void)n_in; (void)out_size; (void)ws_size;
}
